// MultiTaskGNN_66168266162313
// MI455X (gfx1250) — compile-verified, matrix operand order checked by reading, not run on hardware
//
#include <hip/hip_runtime.h>
#include <hip/hip_bf16.h>

// ---------------------------------------------------------------------------
// MultiTask GATv2 GNN for gfx1250 (MI455X).
// Dense GEMMs: v_wmma_f32_16x16x32_bf16, A staged via LDS (coalesced b128),
// B pre-swizzled in global memory into the per-lane WMMA fragment layout
// (L2-resident weights -> direct 32B/lane fragment loads, no LDS transpose).
// Edge phase: warp-per-edge b128 gathers + wave32 shfl reductions + atomics.
// ---------------------------------------------------------------------------

typedef __attribute__((ext_vector_type(16))) __bf16 v16bf;
typedef __attribute__((ext_vector_type(8)))  float  v8f;

#define IN_C  128
#define EMB   256
#define HEADS 4
#define HID   64
#define NLAYERS 3
#define BN_EPS 1e-5f

// ------------------------------- utilities --------------------------------

__global__ void k_fill_f32(float* __restrict__ p, size_t n, float v) {
    size_t t = (size_t)blockIdx.x * blockDim.x + threadIdx.x;
    if (t < n) p[t] = v;
}

__global__ void k_f32_to_bf16(const float* __restrict__ in, __bf16* __restrict__ out, size_t n) {
    size_t t = (size_t)blockIdx.x * blockDim.x + threadIdx.x;
    if (t < n) out[t] = (__bf16)in[t];
}

__device__ __forceinline__ void atomicMaxFloat(float* addr, float val) {
    // int-punning float max: positives compare as ints, negatives as unsigned-min
    if (val >= 0.0f) atomicMax((int*)addr, __float_as_int(val));
    else             atomicMin((unsigned int*)addr, (unsigned int)__float_as_int(val));
}

// --------------------- weight pre-swizzle (f32 -> bf16) -------------------
// Converts W[K,Nn] row-major f32 into per-lane WMMA B-fragment layout:
//   frag index = ((nt*(K/32) + kc)*32 + lane), 16 bf16 per fragment.
// Per ISA 16-bit layout: lanes 0-15 hold K {0..7,16..23}, lanes 16-31 hold
// K {8..15,24..31} of the 32-deep chunk; N = nt*16 + (lane&15).
// One thread per fragment -> one contiguous 32B store.

__global__ void k_swizzle_w(const float* __restrict__ W, __bf16* __restrict__ out,
                            int K, int Nn)
{
    const int KC = K >> 5;
    const int total = (Nn >> 4) * KC * 32;
    const int t = (int)((size_t)blockIdx.x * blockDim.x + threadIdx.x);
    if (t >= total) return;
    const int lane = t & 31;
    const int rest = t >> 5;
    const int kc = rest % KC;
    const int nt = rest / KC;
    const int col = (nt << 4) + (lane & 15);
    const int kbase = (kc << 5) + ((lane >> 4) << 3);

    __bf16 frag[16];
#pragma unroll
    for (int i = 0; i < 8; ++i) {
        frag[i]     = (__bf16)W[(size_t)(kbase + i) * Nn + col];
        frag[i + 8] = (__bf16)W[(size_t)(kbase + 16 + i) * Nn + col];
    }
    *(v16bf*)(out + (size_t)t * 16) = *(const v16bf*)frag;
}

// ------------------------- WMMA bf16 GEMM (64x64 tile) --------------------
// C[M,Nn] = A[M,K] (bf16 row-major) @ W (pre-swizzled bf16) [+ bias[Nn]]
// Block: 256 threads = 8 waves. wave -> M-subtile (wave&3)*16, N-group
// (wave>>2)*32; each wave computes two 16x16 C tiles sharing one A fragment.
// K multiple of 32, Nn multiple of 64.

__global__ __launch_bounds__(256) void k_gemm_bf16_wmma(
    const __bf16* __restrict__ A, const __bf16* __restrict__ Bsw,
    const float* __restrict__ bias, float* __restrict__ C,
    int M, int Nn, int K)
{
    // stride 48 elements (96B) keeps 16B alignment for b128 LDS accesses
    __shared__ __bf16 sA[64][48];   // [m][k]

    const int tid  = threadIdx.x;
    const int lane = tid & 31;
    const int wave = tid >> 5;
    const int wm   = (wave & 3) << 4;       // 0,16,32,48
    const int wg   = (wave >> 2);           // N group 0/1 (32 cols each)
    const int row  = lane & 15;
    const int kh   = (lane >> 4) & 1;       // lane half selects K group
    const int m0   = blockIdx.x * 64;
    const int n0   = blockIdx.y * 64;

    const int KC  = K >> 5;
    const int nt0 = (n0 >> 4) + (wg << 1);  // first 16-wide N tile for wave
    const __bf16* bp0 = Bsw + ((size_t)nt0       * KC) * 32 * 16 + (size_t)lane * 16;
    const __bf16* bp1 = Bsw + ((size_t)(nt0 + 1) * KC) * 32 * 16 + (size_t)lane * 16;

    const int ar  = tid >> 2;               // 0..63
    const int ac8 = (tid & 3) << 3;         // 0,8,16,24

    v8f c0 = {};
    v8f c1 = {};

    for (int kc = 0; kc < KC; ++kc) {
        // --- stage A tile 64x32 (uint4 = 8 bf16 per thread, coalesced) ---
        uint4 av = make_uint4(0u, 0u, 0u, 0u);
        const int gr = m0 + ar;
        if (gr < M) av = *(const uint4*)(A + (size_t)gr * K + (kc << 5) + ac8);
        *(uint4*)&sA[ar][ac8] = av;
        __syncthreads();

        // --- A fragment: 2 contiguous 16B LDS reads per ISA 16-bit layout
        v16bf a, b0, b1;
        const __bf16* pa = &sA[wm + row][kh << 3];
#pragma unroll
        for (int i = 0; i < 8; ++i) {
            a[i]     = pa[i];
            a[i + 8] = pa[16 + i];
        }
        // --- B fragments: direct 32B/lane loads from pre-swizzled weights
        b0 = *(const v16bf*)(bp0 + (size_t)kc * 32 * 16);
        b1 = *(const v16bf*)(bp1 + (size_t)kc * 32 * 16);

        c0 = __builtin_amdgcn_wmma_f32_16x16x32_bf16(
                 false, a, false, b0, (short)0, c0, false, false);
        c1 = __builtin_amdgcn_wmma_f32_16x16x32_bf16(
                 false, a, false, b1, (short)0, c1, false, false);
        __syncthreads();
    }

    // --- store: VGPR r -> M = r (lanes<16) / r+8 (lanes>=16); N = lane&15
    const int mbase = m0 + wm + (kh << 3);
    const int na    = n0 + (wg << 5) + row;
    const int nb    = na + 16;
    const float bva = bias ? bias[na] : 0.0f;
    const float bvb = bias ? bias[nb] : 0.0f;
#pragma unroll
    for (int r = 0; r < 8; ++r) {
        const int m = mbase + r;
        if (m < M) {
            C[(size_t)m * Nn + na] = c0[r] + bva;
            C[(size_t)m * Nn + nb] = c1[r] + bvb;
        }
    }
}

// -------------------- BN(eval) + ELU (+bias, +residual) -------------------
// Also maintains the bf16 shadow of h for the next WMMA GEMM.

__global__ void k_bn_elu(const float* __restrict__ in,
                         const float* __restrict__ prebias,   // may be null
                         const float* __restrict__ g, const float* __restrict__ b,
                         const float* __restrict__ mean, const float* __restrict__ var,
                         const float* __restrict__ resid,     // may be null
                         float* __restrict__ h, __bf16* __restrict__ hb,
                         size_t total)
{
    size_t t = (size_t)blockIdx.x * blockDim.x + threadIdx.x;
    if (t >= total) return;
    const int c = (int)(t & (EMB - 1));
    float v = in[t];
    if (prebias) v += prebias[c];
    v = (v - mean[c]) * (g[c] * rsqrtf(var[c] + BN_EPS)) + b[c];
    v = v > 0.0f ? v : expm1f(v);                 // ELU
    if (resid) v += resid[t];
    h[t]  = v;
    hb[t] = (__bf16)v;
}

// --------------------------- edge kernels ---------------------------------
// Warp (wave32) per edge; lane owns 8 contiguous channels; lanes 8k..8k+7
// all belong to head k, so per-head reduction = shfl_xor within 8 lanes.

__device__ __forceinline__ void edge_ids(int e, const int* __restrict__ ei,
                                         int NE, int& s, int& d) {
    if (e < NE) { s = ei[e]; d = ei[NE + e]; }
    else        { s = e - NE; d = s; }            // self loop
}

__global__ void k_edge_logits(const float* __restrict__ xl, const float* __restrict__ xr,
                              const int* __restrict__ ei, const float* __restrict__ att,
                              float* __restrict__ elog, float* __restrict__ mmax,
                              int NE, int Etot)
{
    const int gw   = (int)(((size_t)blockIdx.x * blockDim.x + threadIdx.x) >> 5);
    const int lane = threadIdx.x & 31;
    if (gw >= Etot) return;
    int s, d; edge_ids(gw, ei, NE, s, d);

    const float4* pl = (const float4*)(xl + (size_t)s * EMB);
    const float4* pr = (const float4*)(xr + (size_t)d * EMB);
    const int head = lane >> 3;
    const int cc   = (lane & 7) << 3;             // channel-in-head base
    const float* ah = att + head * HID;

    float partial = 0.0f;
#pragma unroll
    for (int j = 0; j < 2; ++j) {
        const float4 l4 = pl[2 * lane + j];
        const float4 r4 = pr[2 * lane + j];
        float s0 = l4.x + r4.x; s0 = s0 > 0.0f ? s0 : 0.2f * s0;
        float s1 = l4.y + r4.y; s1 = s1 > 0.0f ? s1 : 0.2f * s1;
        float s2 = l4.z + r4.z; s2 = s2 > 0.0f ? s2 : 0.2f * s2;
        float s3 = l4.w + r4.w; s3 = s3 > 0.0f ? s3 : 0.2f * s3;
        partial += s0 * ah[cc + 4 * j + 0] + s1 * ah[cc + 4 * j + 1]
                 + s2 * ah[cc + 4 * j + 2] + s3 * ah[cc + 4 * j + 3];
    }
#pragma unroll
    for (int off = 1; off < 8; off <<= 1)
        partial += __shfl_xor(partial, off, 32);

    if ((lane & 7) == 0) {
        elog[(size_t)gw * HEADS + head] = partial;
        atomicMaxFloat(&mmax[(size_t)d * HEADS + head], partial);
    }
}

__global__ void k_edge_softmax(float* __restrict__ elog,
                               const float* __restrict__ mmax,
                               float* __restrict__ denom,
                               const int* __restrict__ ei, int NE, int Etot)
{
    const size_t t = (size_t)blockIdx.x * blockDim.x + threadIdx.x;
    if (t >= (size_t)Etot * HEADS) return;
    const int e = (int)(t >> 2), hh = (int)(t & 3);
    int s, d; edge_ids(e, ei, NE, s, d); (void)s;
    const float p = __expf(elog[t] - mmax[(size_t)d * HEADS + hh]);
    elog[t] = p;
    atomicAdd(&denom[(size_t)d * HEADS + hh], p);
}

__global__ void k_edge_aggregate(const float* __restrict__ xl,
                                 const float* __restrict__ elog,
                                 const float* __restrict__ denom,
                                 const int* __restrict__ ei,
                                 float* __restrict__ agg, int NE, int Etot)
{
    const int gw   = (int)(((size_t)blockIdx.x * blockDim.x + threadIdx.x) >> 5);
    const int lane = threadIdx.x & 31;
    if (gw >= Etot) return;
    int s, d; edge_ids(gw, ei, NE, s, d);
    const int head = lane >> 3;
    const float alpha = elog[(size_t)gw * HEADS + head]
                      / denom[(size_t)d * HEADS + head];
    const float4* pl = (const float4*)(xl + (size_t)s * EMB);
    float* po = agg + (size_t)d * EMB + lane * 8;
#pragma unroll
    for (int j = 0; j < 2; ++j) {
        const float4 l4 = pl[2 * lane + j];
        atomicAdd(po + 4 * j + 0, alpha * l4.x);
        atomicAdd(po + 4 * j + 1, alpha * l4.y);
        atomicAdd(po + 4 * j + 2, alpha * l4.z);
        atomicAdd(po + 4 * j + 3, alpha * l4.w);
    }
}

// ------------------------------ head kernel -------------------------------
// preds[N,4] = h @ head_W.T + head_b ; Nc=4 too small for WMMA, VALU kernel.
// Output layout: pred_t at d_out[t*N + i].

__global__ void k_head(const float* __restrict__ h, const float* __restrict__ W,
                       const float* __restrict__ bvec, float* __restrict__ out, int n)
{
    const int i = (int)((size_t)blockIdx.x * blockDim.x + threadIdx.x);
    if (i >= n) return;
    const float4* ph = (const float4*)(h + (size_t)i * EMB);
    float acc0 = bvec[0], acc1 = bvec[1], acc2 = bvec[2], acc3 = bvec[3];
    const float4* w0 = (const float4*)(W + 0 * EMB);
    const float4* w1 = (const float4*)(W + 1 * EMB);
    const float4* w2 = (const float4*)(W + 2 * EMB);
    const float4* w3 = (const float4*)(W + 3 * EMB);
    for (int j = 0; j < EMB / 4; ++j) {
        const float4 v = ph[j];
        float4 w;
        w = w0[j]; acc0 += v.x * w.x + v.y * w.y + v.z * w.z + v.w * w.w;
        w = w1[j]; acc1 += v.x * w.x + v.y * w.y + v.z * w.z + v.w * w.w;
        w = w2[j]; acc2 += v.x * w.x + v.y * w.y + v.z * w.z + v.w * w.w;
        w = w3[j]; acc3 += v.x * w.x + v.y * w.y + v.z * w.z + v.w * w.w;
    }
    out[0 * (size_t)n + i] = acc0;
    out[1 * (size_t)n + i] = acc1;
    out[2 * (size_t)n + i] = acc2;
    out[3 * (size_t)n + i] = acc3;
}

// ------------------------------ launch ------------------------------------

static inline size_t align256(size_t x) { return (x + 255) & ~(size_t)255; }

extern "C" void kernel_launch(void* const* d_in, const int* in_sizes, int n_in,
                              void* d_out, int out_size, void* d_ws, size_t ws_size,
                              hipStream_t stream) {
    (void)n_in; (void)out_size; (void)ws_size;

    const float* x        = (const float*)d_in[0];
    const int*   ei       = (const int*)  d_in[1];
    const float* proj_W   = (const float*)d_in[2];
    const float* proj_b   = (const float*)d_in[3];
    const float* ibn_g    = (const float*)d_in[4];
    const float* ibn_b    = (const float*)d_in[5];
    const float* ibn_m    = (const float*)d_in[6];
    const float* ibn_v    = (const float*)d_in[7];
    const float* Wl       = (const float*)d_in[8];
    const float* Wr       = (const float*)d_in[9];
    const float* att      = (const float*)d_in[10];
    const float* conv_b   = (const float*)d_in[11];
    const float* bn_g     = (const float*)d_in[12];
    const float* bn_b     = (const float*)d_in[13];
    const float* bn_m     = (const float*)d_in[14];
    const float* bn_v     = (const float*)d_in[15];
    const float* head_W   = (const float*)d_in[16];
    const float* head_b   = (const float*)d_in[17];
    const float* dec_W    = (const float*)d_in[18];
    const float* dec_b    = (const float*)d_in[19];

    const int N    = in_sizes[0] / IN_C;
    const int NE   = in_sizes[1] / 2;
    const int Etot = NE + N;

    // ---- output layout: [4 preds | h (N*EMB) | recon (N*IN_C)] -----------
    float* out   = (float*)d_out;
    float* h     = out + (size_t)4 * N;            // final h lives in d_out
    float* recon = out + (size_t)4 * N + (size_t)N * EMB;

    // ---- workspace carve ------------------------------------------------
    char* ws = (char*)d_ws;
    size_t off = 0;
    __bf16* x_bf   = (__bf16*)(ws + off); off += align256((size_t)N * IN_C * 2);
    __bf16* h_bf   = (__bf16*)(ws + off); off += align256((size_t)N * EMB * 2);
    float*  xl     = (float*) (ws + off); off += align256((size_t)N * EMB * 4);
    float*  xr     = (float*) (ws + off); off += align256((size_t)N * EMB * 4);
    float*  agg    = (float*) (ws + off); off += align256((size_t)N * EMB * 4);
    float*  mmax   = (float*) (ws + off); off += align256((size_t)N * HEADS * 4);
    float*  denom  = (float*) (ws + off); off += align256((size_t)N * HEADS * 4);
    float*  elog   = (float*) (ws + off); off += align256((size_t)Etot * HEADS * 4);
    __bf16* projWs = (__bf16*)(ws + off); off += align256((size_t)IN_C * EMB * 2);
    __bf16* Wls    = (__bf16*)(ws + off); off += align256((size_t)NLAYERS * EMB * EMB * 2);
    __bf16* Wrs    = (__bf16*)(ws + off); off += align256((size_t)NLAYERS * EMB * EMB * 2);
    __bf16* decWs  = (__bf16*)(ws + off); off += align256((size_t)EMB * IN_C * 2);

    const int TPB = 256;
    auto nb = [](size_t n, int tpb) { return (unsigned)((n + tpb - 1) / tpb); };

    // ---- activations -> bf16 ; weights -> swizzled bf16 fragments -------
    k_f32_to_bf16<<<nb((size_t)N * IN_C, TPB), TPB, 0, stream>>>(x, x_bf, (size_t)N * IN_C);
    k_swizzle_w<<<nb((size_t)IN_C * EMB / 16, TPB), TPB, 0, stream>>>(proj_W, projWs, IN_C, EMB);
    for (int i = 0; i < NLAYERS; ++i) {
        k_swizzle_w<<<nb((size_t)EMB * EMB / 16, TPB), TPB, 0, stream>>>(
            Wl + (size_t)i * EMB * EMB, Wls + (size_t)i * EMB * EMB, EMB, EMB);
        k_swizzle_w<<<nb((size_t)EMB * EMB / 16, TPB), TPB, 0, stream>>>(
            Wr + (size_t)i * EMB * EMB, Wrs + (size_t)i * EMB * EMB, EMB, EMB);
    }
    k_swizzle_w<<<nb((size_t)EMB * IN_C / 16, TPB), TPB, 0, stream>>>(dec_W, decWs, EMB, IN_C);

    const dim3 gemm_grid_emb((N + 63) / 64, EMB / 64);
    const dim3 gemm_grid_dec((N + 63) / 64, IN_C / 64);

    // ---- projection: tmp = x @ proj_W + proj_b ; h0 = elu(bn(tmp)) ------
    k_gemm_bf16_wmma<<<gemm_grid_emb, 256, 0, stream>>>(x_bf, projWs, proj_b, xl, N, EMB, IN_C);
    k_bn_elu<<<nb((size_t)N * EMB, TPB), TPB, 0, stream>>>(
        xl, nullptr, ibn_g, ibn_b, ibn_m, ibn_v, nullptr, h, h_bf, (size_t)N * EMB);

    // ---- 3 GATv2 layers -------------------------------------------------
    const unsigned edge_warps_blocks = nb((size_t)Etot * 32, TPB);
    for (int i = 0; i < NLAYERS; ++i) {
        k_fill_f32<<<nb((size_t)N * HEADS, TPB), TPB, 0, stream>>>(mmax, (size_t)N * HEADS, -__builtin_huge_valf());
        k_fill_f32<<<nb((size_t)N * HEADS, TPB), TPB, 0, stream>>>(denom, (size_t)N * HEADS, 0.0f);
        k_fill_f32<<<nb((size_t)N * EMB, TPB), TPB, 0, stream>>>(agg, (size_t)N * EMB, 0.0f);

        k_gemm_bf16_wmma<<<gemm_grid_emb, 256, 0, stream>>>(
            h_bf, Wls + (size_t)i * EMB * EMB, nullptr, xl, N, EMB, EMB);
        k_gemm_bf16_wmma<<<gemm_grid_emb, 256, 0, stream>>>(
            h_bf, Wrs + (size_t)i * EMB * EMB, nullptr, xr, N, EMB, EMB);

        k_edge_logits<<<edge_warps_blocks, TPB, 0, stream>>>(
            xl, xr, ei, att + (size_t)i * HEADS * HID, elog, mmax, NE, Etot);
        k_edge_softmax<<<nb((size_t)Etot * HEADS, TPB), TPB, 0, stream>>>(
            elog, mmax, denom, ei, NE, Etot);
        k_edge_aggregate<<<edge_warps_blocks, TPB, 0, stream>>>(
            xl, elog, denom, ei, agg, NE, Etot);

        k_bn_elu<<<nb((size_t)N * EMB, TPB), TPB, 0, stream>>>(
            agg, conv_b + (size_t)i * EMB,
            bn_g + (size_t)i * EMB, bn_b + (size_t)i * EMB,
            bn_m + (size_t)i * EMB, bn_v + (size_t)i * EMB,
            h /*residual*/, h, h_bf, (size_t)N * EMB);
    }

    // ---- heads + decoder ------------------------------------------------
    k_head<<<nb((size_t)N, TPB), TPB, 0, stream>>>(h, head_W, head_b, out, N);
    k_gemm_bf16_wmma<<<gemm_grid_dec, 256, 0, stream>>>(h_bf, decWs, dec_b, recon, N, IN_C, EMB);
}
